// DDiTBlock_81518479278607
// MI455X (gfx1250) — matrix operand order repaired and checked by reading, not compiled, not run on hardware
//
#include <hip/hip_runtime.h>
#include <hip/hip_bf16.h>

// ---------------------------------------------------------------------------
// Types for CDNA5 WMMA (wave32): v16bf A/B operands, v8f C/D accumulator
// ---------------------------------------------------------------------------
typedef __attribute__((ext_vector_type(16))) __bf16 v16bf;
typedef __attribute__((ext_vector_type(8)))  __bf16 v8bf;
typedef __attribute__((ext_vector_type(8)))  float  v8f;

#define DIM      768
#define NHEADS   12
#define DH       64
#define SEQ      2048
#define BATCH    2
#define ROWS     (BATCH * SEQ)        // 4096
#define HIDDEN   (4 * DIM)            // 3072
#define ADA6     (6 * DIM)            // 4608

static __device__ __forceinline__ v16bf ld_ab(const __bf16* lo, const __bf16* hi) {
  v8bf a = *(const v8bf*)lo;
  v8bf b = *(const v8bf*)hi;
  v16bf r;
#pragma unroll
  for (int i = 0; i < 8; ++i) { r[i] = a[i]; r[i + 8] = b[i]; }
  return r;
}

static __device__ __forceinline__ float gelu_tanh(float x) {
  float x3 = x * x * x;
  return 0.5f * x * (1.0f + tanhf(0.7978845608028654f * (x + 0.044715f * x3)));
}

// Butterfly xor-shuffle restricted to 16-lane halves (M = 1/2/4/8; permlane16 selects xor the lane by 1/3/7/8, shfl by M).
// Prefer v_permlane16_b32 (VALU, no DS counter traffic); fall back to shfl.
template <int M>
static __device__ __forceinline__ float shfl_xor16(float v) {
#if __has_builtin(__builtin_amdgcn_permlane16)
  unsigned s0, s1;
  if (M == 1)      { s0 = 0x67452301u; s1 = 0xEFCDAB89u; }
  else if (M == 2) { s0 = 0x45670123u; s1 = 0xCDEF89ABu; }
  else if (M == 4) { s0 = 0x01234567u; s1 = 0x89ABCDEFu; }
  else             { s0 = 0xFEDCBA98u; s1 = 0x76543210u; }
  unsigned u = __builtin_bit_cast(unsigned, v);
  u = (unsigned)__builtin_amdgcn_permlane16((int)u, (int)u, s0, s1, false, false);
  return __builtin_bit_cast(float, u);
#else
  return __shfl_xor(v, M, 32);
#endif
}

static __device__ __forceinline__ float rowmax16(float v) {
  v = fmaxf(v, shfl_xor16<8>(v));
  v = fmaxf(v, shfl_xor16<4>(v));
  v = fmaxf(v, shfl_xor16<2>(v));
  v = fmaxf(v, shfl_xor16<1>(v));
  return v;
}
static __device__ __forceinline__ float rowsum16(float v) {
  v += shfl_xor16<8>(v);
  v += shfl_xor16<4>(v);
  v += shfl_xor16<2>(v);
  v += shfl_xor16<1>(v);
  return v;
}

// ---------------------------------------------------------------------------
// 1) adaLN: ada[b, 6*768] = c[b] @ w_ada^T + b_ada.  One wave per output row.
// ---------------------------------------------------------------------------
__global__ __launch_bounds__(256) void ada_kernel(const float* __restrict__ c,
                                                  const float* __restrict__ w,
                                                  const float* __restrict__ bias,
                                                  float* __restrict__ out) {
  int row  = blockIdx.x * 8 + (threadIdx.x >> 5);   // 0 .. 9215
  int lane = threadIdx.x & 31;
  int b = row / ADA6;
  int r = row - b * ADA6;
  const float* cv = c + b * DIM;
  const float* wr = w + (size_t)r * DIM;
  float s = 0.f;
  for (int k = lane; k < DIM; k += 32) s += cv[k] * wr[k];
#pragma unroll
  for (int off = 16; off >= 1; off >>= 1) s += __shfl_xor(s, off, 32);
  if (lane == 0) out[row] = s + bias[r];
}

// ---------------------------------------------------------------------------
// 2) f32 -> bf16 conversion
// ---------------------------------------------------------------------------
__global__ __launch_bounds__(256) void cvt_bf16(const float* __restrict__ in,
                                                __bf16* __restrict__ out, int n) {
  int i = blockIdx.x * 256 + threadIdx.x;
  if (i < n) out[i] = (__bf16)in[i];
}

// ---------------------------------------------------------------------------
// 3) LayerNorm + adaLN modulation -> bf16 activations. One block per row.
// ---------------------------------------------------------------------------
__global__ __launch_bounds__(256) void ln_mod_kernel(const float* __restrict__ x,
                                                     const float* __restrict__ lnw,
                                                     const float* __restrict__ ada,
                                                     int shiftIdx, int scaleIdx,
                                                     __bf16* __restrict__ out) {
  int row = blockIdx.x;
  int t   = threadIdx.x;
  int b   = row >> 11;
  const float* xr = x + (size_t)row * DIM;
  float v0 = xr[t], v1 = xr[t + 256], v2 = xr[t + 512];
  __shared__ float ssum[256], ssq[256];
  ssum[t] = v0 + v1 + v2;
  ssq[t]  = v0 * v0 + v1 * v1 + v2 * v2;
  __syncthreads();
  for (int o = 128; o >= 1; o >>= 1) {
    if (t < o) { ssum[t] += ssum[t + o]; ssq[t] += ssq[t + o]; }
    __syncthreads();
  }
  float mu  = ssum[0] * (1.0f / DIM);
  float var = ssq[0] * (1.0f / DIM) - mu * mu;
  float inv = rsqrtf(var + 1e-5f);
  const float* ab = ada + b * ADA6;
#pragma unroll
  for (int i = 0; i < 3; ++i) {
    int col  = t + i * 256;
    float xv = (i == 0) ? v0 : (i == 1) ? v1 : v2;
    float sc = 1.0f + ab[scaleIdx * DIM + col];
    float sh = ab[shiftIdx * DIM + col];
    out[(size_t)row * DIM + col] = (__bf16)(((xv - mu) * inv) * lnw[col] * sc + sh);
  }
}

// ---------------------------------------------------------------------------
// 4) Generic WMMA GEMM: C[M,N] = A[M,K] @ W[N,K]^T  (A,W bf16, acc f32)
//    Block = 4 waves -> 128x128 tile; each wave 64x64 (4x4 WMMA 16x16x32).
//    8 fragment loads per 16 WMMAs -> 2x register-level reuse.
//    MODE 0: out bf16 raw (qkv)
//    MODE 1: outF = skip + gate*acc              (attn out-proj)
//    MODE 2: out bf16 = gelu(acc + bias)         (mlp1)
//    MODE 3: outF = skip + gate*(acc + bias)     (mlp2 -> final output)
// ---------------------------------------------------------------------------
template <int MODE>
__global__ __launch_bounds__(128)
void gemm_bf16(const __bf16* __restrict__ A, const __bf16* __restrict__ W,
               int N, int K,
               const float* __restrict__ ada, int gateIdx,
               const float* __restrict__ bias,
               const float* __restrict__ skip,
               float* __restrict__ outF, __bf16* __restrict__ outB) {
  const int lane = threadIdx.x & 31;
  const int w    = threadIdx.x >> 5;          // 0..3
  const int wr   = w & 1, wc = w >> 1;        // 2x2 wave grid
  const int r0   = blockIdx.x * 128 + wr * 64;
  const int c0   = blockIdx.y * 128 + wc * 64;
  const int half = lane >> 4;
  const int l15  = lane & 15;

  v8f acc[4][4] = {};
  for (int k0 = 0; k0 < K; k0 += 32) {
    v16bf a[4], bfr[4];
    int abase = k0 + half * 8;
#pragma unroll
    for (int i = 0; i < 4; ++i) {
      const __bf16* pa = A + (size_t)(r0 + i * 16 + l15) * K + abase;
      a[i] = ld_ab(pa, pa + 16);
    }
    int bbase = k0 + half * 8;
#pragma unroll
    for (int j = 0; j < 4; ++j) {
      const __bf16* pw = W + (size_t)(c0 + j * 16 + l15) * K + bbase;
      bfr[j] = ld_ab(pw, pw + 16);
    }
#pragma unroll
    for (int i = 0; i < 4; ++i)
#pragma unroll
      for (int j = 0; j < 4; ++j)
        acc[i][j] = __builtin_amdgcn_wmma_f32_16x16x32_bf16(
            false, a[i], false, bfr[j], (short)0, acc[i][j], false, false);
  }

#pragma unroll
  for (int i = 0; i < 4; ++i) {
#pragma unroll
    for (int j = 0; j < 4; ++j) {
#pragma unroll
      for (int g = 0; g < 8; ++g) {
        int m = r0 + i * 16 + half * 8 + g;
        int n = c0 + j * 16 + l15;
        float v = acc[i][j][g];
        if constexpr (MODE == 0) {
          outB[(size_t)m * N + n] = (__bf16)v;
        } else if constexpr (MODE == 1) {
          int bb = m >> 11;
          float gate = ada[bb * ADA6 + gateIdx * DIM + n];
          outF[(size_t)m * DIM + n] = skip[(size_t)m * DIM + n] + gate * v;
        } else if constexpr (MODE == 2) {
          outB[(size_t)m * N + n] = (__bf16)gelu_tanh(v + bias[n]);
        } else {
          int bb = m >> 11;
          float gate = ada[bb * ADA6 + gateIdx * DIM + n];
          outF[(size_t)m * DIM + n] =
              skip[(size_t)m * DIM + n] + gate * (v + bias[n]);
        }
      }
    }
  }
}

// ---------------------------------------------------------------------------
// 5) Reorder qkv[m, 2304] -> q[b,h,s,d], k[b,h,s,d], vt[b,h,d,s] (bf16)
// ---------------------------------------------------------------------------
__global__ __launch_bounds__(256) void reorder_qkv(const __bf16* __restrict__ qkv,
                                                   __bf16* __restrict__ q,
                                                   __bf16* __restrict__ k,
                                                   __bf16* __restrict__ vt) {
  int idx = blockIdx.x * 256 + threadIdx.x;
  if (idx >= ROWS * 3 * DIM) return;
  int m = idx / (3 * DIM);
  int cidx = idx - m * (3 * DIM);
  int which = cidx / DIM;
  int r  = cidx - which * DIM;
  int h  = r >> 6;
  int dh = r & 63;
  int b  = m >> 11;
  int s  = m & 2047;
  int bh = b * NHEADS + h;
  __bf16 val = qkv[idx];
  if (which == 0)      q [((size_t)bh * SEQ + s) * DH + dh] = val;
  else if (which == 1) k [((size_t)bh * SEQ + s) * DH + dh] = val;
  else                 vt[((size_t)bh * DH + dh) * SEQ + s] = val;
}

// ---------------------------------------------------------------------------
// 6) Flash attention: one wave per 16-row Q tile; 64 KV columns per step
//    (4 score tiles share one butterfly reduction). WMMA for S and P@V.
// ---------------------------------------------------------------------------
__global__ __launch_bounds__(32) void attn_kernel(const __bf16* __restrict__ qh,
                                                  const __bf16* __restrict__ kh,
                                                  const __bf16* __restrict__ vth,
                                                  __bf16* __restrict__ obuf) {
  const int lane = threadIdx.x;
  const int bid  = blockIdx.x;            // 0 .. B*H*(S/16)-1
  const int tile = bid & 127;
  const int bh   = bid >> 7;
  const int h    = bh % NHEADS, b = bh / NHEADS;
  const int half = lane >> 4, l15 = lane & 15;
  const __bf16* Q  = qh  + (size_t)bh * SEQ * DH;
  const __bf16* Kp = kh  + (size_t)bh * SEQ * DH;
  const __bf16* Vt = vth + (size_t)bh * DH * SEQ;
  const int q0 = tile * 16;

  // Q A-frags for the two K-dim steps (Dh = 64 = 2 x 32)
  const __bf16* paq = Q + (size_t)(q0 + l15) * DH + half * 8;
  v16bf aq0 = ld_ab(paq, paq + 16);
  v16bf aq1 = ld_ab(paq + 32, paq + 48);

  v8f acc[4] = {};                        // O tile: 16 x 64
  float mrun[8], lrun[8];
#pragma unroll
  for (int g = 0; g < 8; ++g) { mrun[g] = -3.0e38f; lrun[g] = 0.f; }

  __shared__ __align__(64) __bf16 pl[16 * 64];   // P tile 16x64 bf16

  for (int kv0 = 0; kv0 < SEQ; kv0 += 64) {
    // ---- scores S = Q K^T (four 16-wide column tiles) ----
    v8f s[4];
#pragma unroll
    for (int t = 0; t < 4; ++t) {
      const __bf16* pk = Kp + (size_t)(kv0 + t * 16 + l15) * DH + half * 8;
      v16bf bk0 = ld_ab(pk, pk + 16);
      v16bf bk1 = ld_ab(pk + 32, pk + 48);
      v8f z = {};
      z = __builtin_amdgcn_wmma_f32_16x16x32_bf16(false, aq0, false, bk0, (short)0, z, false, false);
      z = __builtin_amdgcn_wmma_f32_16x16x32_bf16(false, aq1, false, bk1, (short)0, z, false, false);
      s[t] = z;
    }

    // ---- online softmax: one butterfly per row per 64 columns ----
#pragma unroll
    for (int g = 0; g < 8; ++g) {
      float r0 = s[0][g] * 0.125f, r1 = s[1][g] * 0.125f;
      float r2 = s[2][g] * 0.125f, r3 = s[3][g] * 0.125f;
      float mx = rowmax16(fmaxf(fmaxf(r0, r1), fmaxf(r2, r3)));
      float mnew = fmaxf(mrun[g], mx);
      float corr = __expf(mrun[g] - mnew);
      float e0 = __expf(r0 - mnew);
      float e1 = __expf(r1 - mnew);
      float e2 = __expf(r2 - mnew);
      float e3 = __expf(r3 - mnew);
      float rs = rowsum16(e0 + e1 + e2 + e3);
      lrun[g] = lrun[g] * corr + rs;
      mrun[g] = mnew;
      // stash exp'ed probs back into s[] (C layout) for LDS staging
      s[0][g] = e0; s[1][g] = e1; s[2][g] = e2; s[3][g] = e3;
      acc[0][g] *= corr; acc[1][g] *= corr; acc[2][g] *= corr; acc[3][g] *= corr;
    }

    // ---- re-layout P (C-layout f32) -> A-frag bf16 via LDS ----
#pragma unroll
    for (int g = 0; g < 8; ++g) {
      int mrow = g + half * 8;
#pragma unroll
      for (int t = 0; t < 4; ++t)
        pl[mrow * 64 + t * 16 + l15] = (__bf16)s[t][g];
    }
    __syncthreads();
    const __bf16* pp = &pl[l15 * 64 + half * 8];
    v16bf ap0 = ld_ab(pp,      pp + 16);   // P columns kv0 .. kv0+31
    v16bf ap1 = ld_ab(pp + 32, pp + 48);   // P columns kv0+32 .. kv0+63

    // ---- O += P @ V  (V B-frags contiguous thanks to Vt[d, s]) ----
#pragma unroll
    for (int nt = 0; nt < 4; ++nt) {
      const __bf16* pv = Vt + (size_t)(nt * 16 + l15) * SEQ + kv0 + half * 8;
      v16bf bv0 = ld_ab(pv, pv + 16);
      v16bf bv1 = ld_ab(pv + 32, pv + 48);
      acc[nt] = __builtin_amdgcn_wmma_f32_16x16x32_bf16(false, ap0, false, bv0, (short)0, acc[nt], false, false);
      acc[nt] = __builtin_amdgcn_wmma_f32_16x16x32_bf16(false, ap1, false, bv1, (short)0, acc[nt], false, false);
    }
    __syncthreads();
  }

  // ---- normalize and write o[b, s, h*64 + d] as bf16 ----
#pragma unroll
  for (int g = 0; g < 8; ++g) {
    float inv = 1.0f / lrun[g];
    int s = q0 + g + half * 8;
    size_t base = ((size_t)(b * SEQ + s)) * DIM + h * DH + l15;
    obuf[base]      = (__bf16)(acc[0][g] * inv);
    obuf[base + 16] = (__bf16)(acc[1][g] * inv);
    obuf[base + 32] = (__bf16)(acc[2][g] * inv);
    obuf[base + 48] = (__bf16)(acc[3][g] * inv);
  }
}

// ---------------------------------------------------------------------------
// Launcher
// ---------------------------------------------------------------------------
extern "C" void kernel_launch(void* const* d_in, const int* in_sizes, int n_in,
                              void* d_out, int out_size, void* d_ws, size_t ws_size,
                              hipStream_t stream) {
  const float* x       = (const float*)d_in[0];
  // d_in[1]=cos, d_in[2]=sin : unused by the reference math
  const float* c       = (const float*)d_in[3];
  const float* norm1_w = (const float*)d_in[4];
  const float* w_qkv   = (const float*)d_in[5];
  const float* w_out   = (const float*)d_in[6];
  const float* norm2_w = (const float*)d_in[7];
  const float* w_mlp1  = (const float*)d_in[8];
  const float* b_mlp1  = (const float*)d_in[9];
  const float* w_mlp2  = (const float*)d_in[10];
  const float* b_mlp2  = (const float*)d_in[11];
  const float* w_ada   = (const float*)d_in[12];
  const float* b_ada   = (const float*)d_in[13];
  float* out = (float*)d_out;

  // ---- workspace carve-up (all sizes 256B-aligned) ----
  char* ws = (char*)d_ws;
  size_t off = 0;
  auto carve = [&](size_t bytes) { char* p = ws + off; off += (bytes + 255) & ~(size_t)255; return p; };

  float*  ada    = (float*)carve((size_t)BATCH * ADA6 * 4);
  __bf16* wqkv_b = (__bf16*)carve((size_t)3 * DIM * DIM * 2);
  __bf16* wout_b = (__bf16*)carve((size_t)DIM * DIM * 2);
  __bf16* wm1_b  = (__bf16*)carve((size_t)HIDDEN * DIM * 2);
  __bf16* wm2_b  = (__bf16*)carve((size_t)DIM * HIDDEN * 2);
  __bf16* xmod   = (__bf16*)carve((size_t)ROWS * DIM * 2);      // reused for mod1/mod2
  __bf16* qkvo   = (__bf16*)carve((size_t)ROWS * 3 * DIM * 2);
  __bf16* qbuf   = (__bf16*)carve((size_t)ROWS * DIM * 2);
  __bf16* kbuf   = (__bf16*)carve((size_t)ROWS * DIM * 2);
  __bf16* vtbuf  = (__bf16*)carve((size_t)ROWS * DIM * 2);
  __bf16* obuf   = (__bf16*)carve((size_t)ROWS * DIM * 2);
  float*  x1     = (float*)carve((size_t)ROWS * DIM * 4);
  __bf16* hbuf   = (__bf16*)carve((size_t)ROWS * HIDDEN * 2);

  // 1) adaLN vector
  ada_kernel<<<(BATCH * ADA6) / 8, 256, 0, stream>>>(c, w_ada, b_ada, ada);

  // 2) weights -> bf16
  cvt_bf16<<<(3 * DIM * DIM + 255) / 256, 256, 0, stream>>>(w_qkv, wqkv_b, 3 * DIM * DIM);
  cvt_bf16<<<(DIM * DIM + 255) / 256, 256, 0, stream>>>(w_out, wout_b, DIM * DIM);
  cvt_bf16<<<(HIDDEN * DIM + 255) / 256, 256, 0, stream>>>(w_mlp1, wm1_b, HIDDEN * DIM);
  cvt_bf16<<<(DIM * HIDDEN + 255) / 256, 256, 0, stream>>>(w_mlp2, wm2_b, DIM * HIDDEN);

  // 3) LN1 + modulate (shift=ada[0], scale=ada[1])
  ln_mod_kernel<<<ROWS, 256, 0, stream>>>(x, norm1_w, ada, 0, 1, xmod);

  // 4) QKV projection: [4096,768] @ [2304,768]^T
  gemm_bf16<0><<<dim3(ROWS / 128, (3 * DIM) / 128), 128, 0, stream>>>(
      xmod, wqkv_b, 3 * DIM, DIM, nullptr, 0, nullptr, nullptr, nullptr, qkvo);

  // 5) reorder into head-major q/k and transposed v
  reorder_qkv<<<(ROWS * 3 * DIM + 255) / 256, 256, 0, stream>>>(qkvo, qbuf, kbuf, vtbuf);

  // 6) attention (one wave per 16-row Q tile)
  attn_kernel<<<BATCH * NHEADS * (SEQ / 16), 32, 0, stream>>>(qbuf, kbuf, vtbuf, obuf);

  // 7) out-projection + gate_msa residual -> x1 (f32)
  gemm_bf16<1><<<dim3(ROWS / 128, DIM / 128), 128, 0, stream>>>(
      obuf, wout_b, DIM, DIM, ada, 2, nullptr, x, x1, nullptr);

  // 8) LN2 + modulate (shift=ada[3], scale=ada[4])
  ln_mod_kernel<<<ROWS, 256, 0, stream>>>(x1, norm2_w, ada, 3, 4, xmod);

  // 9) MLP1 + bias + gelu
  gemm_bf16<2><<<dim3(ROWS / 128, HIDDEN / 128), 128, 0, stream>>>(
      xmod, wm1_b, HIDDEN, DIM, nullptr, 0, b_mlp1, nullptr, nullptr, hbuf);

  // 10) MLP2 + bias + gate_mlp residual -> final output (f32)
  gemm_bf16<3><<<dim3(ROWS / 128, DIM / 128), 128, 0, stream>>>(
      hbuf, wm2_b, DIM, HIDDEN, ada, 5, b_mlp2, x1, out, nullptr);

  (void)in_sizes; (void)n_in; (void)out_size; (void)ws_size;
}
